// ExpandingLinear_59004260712486
// MI455X (gfx1250) — matrix operand order repaired and checked by reading, not compiled, not run on hardware
//
#include <hip/hip_runtime.h>
#include <stdint.h>

#define DEVINL __device__ __forceinline__

typedef __attribute__((ext_vector_type(16))) __bf16 v16bf;
typedef __attribute__((ext_vector_type(8)))  __bf16 v8bf;
typedef __attribute__((ext_vector_type(8)))  float  v8f;

// ---------------- problem constants (match reference) ----------------
constexpr int B_    = 2048;
constexpr int F_IN  = 1024;
constexpr int R_EMB = 1024;
constexpr int F_OUT = 1024;
constexpr int F_MID = F_IN + R_EMB;   // 2048
constexpr int NNZ_E = 65536;
constexpr int NNZ_W = 131072;
constexpr int NNZ_B = 512;

// ---------------- bf16 split helpers (no __bf16 arithmetic) ----------------
DEVINL unsigned short f2bf(float f) {            // round-to-nearest-even fp32->bf16
  unsigned u = __float_as_uint(f);
  unsigned r = (u + 0x7FFFu + ((u >> 16) & 1u)) >> 16;
  return (unsigned short)r;
}
DEVINL float bf2f(unsigned short h) { return __uint_as_float(((unsigned)h) << 16); }

DEVINL v8f wmma_bf16(v16bf a, v16bf b, v8f c) {
  // 8 args: (neg_a, A, neg_b, B, c_mod, C, reuse_a, reuse_b)
  return __builtin_amdgcn_wmma_f32_16x16x32_bf16(false, a, false, b, (short)0, c,
                                                 false, false);
}

// A fragment: 16x32 bf16. Lane l holds row m0+(l&15); K chunks per ISA layout:
//   VGPR0-3 : K = (l>>4)*8 .. +7     VGPR4-7 : K = (l>>4)*8+16 .. +23
DEVINL v16bf load_a_frag(const unsigned short* A, int lda, int m0, int k0, int lane) {
  const int row = m0 + (lane & 15);
  const int kh  = (lane >> 4) * 8;
  const v8bf* p0 = reinterpret_cast<const v8bf*>(A + (size_t)row * lda + k0 + kh);
  const v8bf* p1 = reinterpret_cast<const v8bf*>(A + (size_t)row * lda + k0 + kh + 16);
  union { v16bf v; v8bf h[2]; } u;
  u.h[0] = *p0;
  u.h[1] = *p1;
  return u.v;
}

// B fragment: 32x16 bf16. Lane l holds column n0+(l&15); K per ISA layout:
//   elems 0-7 : K = (l>>4)*8 .. +7     elems 8-15 : K = (l>>4)*8+16 .. +23
DEVINL v16bf load_b_frag(const unsigned short* Bt, int ldb, int k0, int n0, int lane) {
  const int n  = n0 + (lane & 15);
  const int kh = (lane >> 4) * 8;
  union { v16bf v; unsigned short s[16]; } u;
#pragma unroll
  for (int s = 0; s < 8; ++s) {
    u.s[s]     = Bt[(size_t)(k0 + kh + s) * ldb + n];
    u.s[8 + s] = Bt[(size_t)(k0 + kh + 16 + s) * ldb + n];
  }
  return u.v;
}

// ---------------- small prep kernels ----------------
__global__ void k_zero_f32(float* p, int n) {
  int i = blockIdx.x * blockDim.x + threadIdx.x;
  if (i < n) p[i] = 0.0f;
}

// dst[cols[i]*ld + rows[i]] += vals[i]  (densify COO weight, transposed)
__global__ void k_scatter_T(const int* __restrict__ rows, const int* __restrict__ cols,
                            const float* __restrict__ vals, float* __restrict__ dst,
                            int ld, int nnz) {
  int i = blockIdx.x * blockDim.x + threadIdx.x;
  if (i < nnz) atomicAdd(dst + (size_t)cols[i] * ld + rows[i], vals[i]);
}

__global__ void k_scatter_bias(const int* __restrict__ idx, const float* __restrict__ vals,
                               float* __restrict__ dst, int nnz) {
  int i = blockIdx.x * blockDim.x + threadIdx.x;
  if (i < nnz) atomicAdd(dst + idx[i], vals[i]);
}

__global__ void k_cvt_hilo(const float* __restrict__ src, unsigned short* __restrict__ hi,
                           unsigned short* __restrict__ lo, int n) {
  int i = blockIdx.x * blockDim.x + threadIdx.x;
  if (i < n) {
    float f = src[i];
    unsigned short h = f2bf(f);
    hi[i] = h;
    lo[i] = f2bf(f - bf2f(h));
  }
}

// x (fp32 [B, F_IN]) -> hi/lo bf16 into left half of x2 [B, F_MID]
__global__ void k_cvt_x(const float* __restrict__ x, unsigned short* __restrict__ x2h,
                        unsigned short* __restrict__ x2l) {
  int i = blockIdx.x * blockDim.x + threadIdx.x;
  if (i < B_ * F_IN) {
    int row = i / F_IN, col = i % F_IN;
    float f = x[i];
    unsigned short h = f2bf(f);
    size_t o = (size_t)row * F_MID + col;
    x2h[o] = h;
    x2l[o] = f2bf(f - bf2f(h));
  }
}

// ---------------- GEMM kernels: wave tile 32M x 64N, K step 32 ----------------
// 3-product bf16 split per tile: hi*hi + hi*lo + lo*hi  (lo*lo ~ 2^-18, dropped)

// GEMM1: h = relu(x[:, :F_IN] @ Wet),  Wet = dense-transposed embed weight
// epilogue writes bf16 hi/lo into right half of x2.
__global__ __launch_bounds__(128)
void k_gemm_embed(const unsigned short* __restrict__ x2h,
                  const unsigned short* __restrict__ x2l,
                  const unsigned short* __restrict__ bth,
                  const unsigned short* __restrict__ btl,
                  unsigned short* __restrict__ outh,
                  unsigned short* __restrict__ outl) {
  const int lane  = threadIdx.x & 31;
  const int gwave = blockIdx.x * (blockDim.x >> 5) + (threadIdx.x >> 5);
  const int nWN   = R_EMB / 64;                 // 16
  const int m0 = (gwave / nWN) * 32;
  const int n0 = (gwave % nWN) * 64;

  v8f acc[2][4] = {};
  for (int k0 = 0; k0 < F_IN; k0 += 32) {
    v16bf ah[2], al[2];
#pragma unroll
    for (int mt = 0; mt < 2; ++mt) {
      ah[mt] = load_a_frag(x2h, F_MID, m0 + 16 * mt, k0, lane);
      al[mt] = load_a_frag(x2l, F_MID, m0 + 16 * mt, k0, lane);
    }
#pragma unroll
    for (int nt = 0; nt < 4; ++nt) {
      v16bf bh = load_b_frag(bth, R_EMB, k0, n0 + 16 * nt, lane);
      v16bf bl = load_b_frag(btl, R_EMB, k0, n0 + 16 * nt, lane);
#pragma unroll
      for (int mt = 0; mt < 2; ++mt) {
        acc[mt][nt] = wmma_bf16(ah[mt], bh, acc[mt][nt]);
        acc[mt][nt] = wmma_bf16(ah[mt], bl, acc[mt][nt]);
        acc[mt][nt] = wmma_bf16(al[mt], bh, acc[mt][nt]);
      }
    }
  }
  // epilogue: relu, bf16-split, store to x2[:, F_IN + n]
#pragma unroll
  for (int mt = 0; mt < 2; ++mt)
#pragma unroll
    for (int nt = 0; nt < 4; ++nt) {
      const int col   = F_IN + n0 + nt * 16 + (lane & 15);
      const int rbase = m0 + mt * 16 + (lane >> 4) * 8;
#pragma unroll
      for (int r = 0; r < 8; ++r) {
        float v = acc[mt][nt][r];
        v = v > 0.0f ? v : 0.0f;
        unsigned short h = f2bf(v);
        size_t o = (size_t)(rbase + r) * F_MID + col;
        outh[o] = h;
        outl[o] = f2bf(v - bf2f(h));
      }
    }
}

// GEMM2: out = x2 @ Wmt + bias
__global__ __launch_bounds__(128)
void k_gemm_main(const unsigned short* __restrict__ x2h,
                 const unsigned short* __restrict__ x2l,
                 const unsigned short* __restrict__ bth,
                 const unsigned short* __restrict__ btl,
                 const float* __restrict__ bias,
                 float* __restrict__ out) {
  const int lane  = threadIdx.x & 31;
  const int gwave = blockIdx.x * (blockDim.x >> 5) + (threadIdx.x >> 5);
  const int nWN   = F_OUT / 64;                 // 16
  const int m0 = (gwave / nWN) * 32;
  const int n0 = (gwave % nWN) * 64;

  v8f acc[2][4] = {};
  for (int k0 = 0; k0 < F_MID; k0 += 32) {
    v16bf ah[2], al[2];
#pragma unroll
    for (int mt = 0; mt < 2; ++mt) {
      ah[mt] = load_a_frag(x2h, F_MID, m0 + 16 * mt, k0, lane);
      al[mt] = load_a_frag(x2l, F_MID, m0 + 16 * mt, k0, lane);
    }
#pragma unroll
    for (int nt = 0; nt < 4; ++nt) {
      v16bf bh = load_b_frag(bth, F_OUT, k0, n0 + 16 * nt, lane);
      v16bf bl = load_b_frag(btl, F_OUT, k0, n0 + 16 * nt, lane);
#pragma unroll
      for (int mt = 0; mt < 2; ++mt) {
        acc[mt][nt] = wmma_bf16(ah[mt], bh, acc[mt][nt]);
        acc[mt][nt] = wmma_bf16(ah[mt], bl, acc[mt][nt]);
        acc[mt][nt] = wmma_bf16(al[mt], bh, acc[mt][nt]);
      }
    }
  }
#pragma unroll
  for (int mt = 0; mt < 2; ++mt)
#pragma unroll
    for (int nt = 0; nt < 4; ++nt) {
      const int col   = n0 + nt * 16 + (lane & 15);
      const int rbase = m0 + mt * 16 + (lane >> 4) * 8;
      const float bv  = bias[col];
#pragma unroll
      for (int r = 0; r < 8; ++r)
        out[(size_t)(rbase + r) * F_OUT + col] = acc[mt][nt][r] + bv;
    }
}

// ---------------- launch ----------------
extern "C" void kernel_launch(void* const* d_in, const int* in_sizes, int n_in,
                              void* d_out, int out_size, void* d_ws, size_t ws_size,
                              hipStream_t stream) {
  const float* x          = (const float*)d_in[0];
  const int*   e_rows     = (const int*)d_in[1];
  const int*   e_cols     = (const int*)d_in[2];
  const float* e_vals     = (const float*)d_in[3];
  const int*   w_rows     = (const int*)d_in[4];
  const int*   w_cols     = (const int*)d_in[5];
  const float* w_vals     = (const float*)d_in[6];
  const int*   bias_idx   = (const int*)d_in[7];
  const float* bias_vals  = (const float*)d_in[8];
  float*       out        = (float*)d_out;

  // ---- workspace layout (bytes), all 256B aligned ----
  char* ws = (char*)d_ws;
  constexpr size_t SZ_WET_F  = (size_t)F_IN * R_EMB * 4;          // 4 MB
  constexpr size_t SZ_WMT_F  = (size_t)F_MID * F_OUT * 4;         // 8 MB
  constexpr size_t SZ_BIAS_F = (size_t)F_OUT * 4;
  constexpr size_t SZ_WET_B  = (size_t)F_IN * R_EMB * 2;          // 2 MB
  constexpr size_t SZ_WMT_B  = (size_t)F_MID * F_OUT * 2;         // 4 MB
  constexpr size_t SZ_X2_B   = (size_t)B_ * F_MID * 2;            // 8 MB

  size_t o = 0;
  float* wet_f  = (float*)(ws + o);          o += SZ_WET_F;
  float* wmt_f  = (float*)(ws + o);          o += SZ_WMT_F;
  float* bias_f = (float*)(ws + o);          o += (SZ_BIAS_F + 255) & ~(size_t)255;
  unsigned short* wet_hi = (unsigned short*)(ws + o); o += SZ_WET_B;
  unsigned short* wet_lo = (unsigned short*)(ws + o); o += SZ_WET_B;
  unsigned short* wmt_hi = (unsigned short*)(ws + o); o += SZ_WMT_B;
  unsigned short* wmt_lo = (unsigned short*)(ws + o); o += SZ_WMT_B;
  unsigned short* x2_hi  = (unsigned short*)(ws + o); o += SZ_X2_B;
  unsigned short* x2_lo  = (unsigned short*)(ws + o); o += SZ_X2_B;
  (void)ws_size; (void)n_in; (void)in_sizes; (void)out_size;

  // 1) zero the contiguous fp32 scratch (Wet, Wmt, bias)
  const int nZero = (int)((SZ_WET_F + SZ_WMT_F + SZ_BIAS_F) / 4) + 64;
  hipLaunchKernelGGL(k_zero_f32, dim3((nZero + 255) / 256), dim3(256), 0, stream,
                     wet_f, nZero);

  // 2) densify (transposed) weights + bias via atomic scatter (handles dup indices)
  hipLaunchKernelGGL(k_scatter_T, dim3(NNZ_E / 256), dim3(256), 0, stream,
                     e_rows, e_cols, e_vals, wet_f, R_EMB, NNZ_E);
  hipLaunchKernelGGL(k_scatter_T, dim3(NNZ_W / 256), dim3(256), 0, stream,
                     w_rows, w_cols, w_vals, wmt_f, F_OUT, NNZ_W);
  hipLaunchKernelGGL(k_scatter_bias, dim3((NNZ_B + 255) / 256), dim3(256), 0, stream,
                     bias_idx, bias_vals, bias_f, NNZ_B);

  // 3) fp32 -> bf16 hi/lo splits
  hipLaunchKernelGGL(k_cvt_hilo, dim3((F_IN * R_EMB) / 256), dim3(256), 0, stream,
                     wet_f, wet_hi, wet_lo, F_IN * R_EMB);
  hipLaunchKernelGGL(k_cvt_hilo, dim3((F_MID * F_OUT) / 256), dim3(256), 0, stream,
                     wmt_f, wmt_hi, wmt_lo, F_MID * F_OUT);
  hipLaunchKernelGGL(k_cvt_x, dim3((B_ * F_IN) / 256), dim3(256), 0, stream,
                     x, x2_hi, x2_lo);

  // 4) GEMM1: relu(x @ We^T) -> x2[:, F_IN:]   (2048x1024x1024, bf16 split WMMA)
  {
    const int waves = (B_ / 32) * (R_EMB / 64);      // 1024 waves
    hipLaunchKernelGGL(k_gemm_embed, dim3(waves / 4), dim3(128), 0, stream,
                       x2_hi, x2_lo, wet_hi, wet_lo, x2_hi, x2_lo);
  }
  // 5) GEMM2: x2 @ Wm^T + bias -> out          (2048x1024x2048)
  {
    const int waves = (B_ / 32) * (F_OUT / 64);      // 1024 waves
    hipLaunchKernelGGL(k_gemm_main, dim3(waves / 4), dim3(128), 0, stream,
                       x2_hi, x2_lo, wmt_hi, wmt_lo, bias_f, out);
  }
}
